// FeatureAttention_39968965657180
// MI455X (gfx1250) — matrix operand order repaired and checked by reading, not compiled, not run on hardware
//
#include <hip/hip_runtime.h>
#include <hip/hip_bf16.h>
#include <hip/hip_fp16.h>

// Problem constants (match reference)
static constexpr int kD = 1024;
static constexpr int kB = 128;

typedef __attribute__((ext_vector_type(16))) _Float16 v16h;
typedef __attribute__((ext_vector_type(8)))  _Float16 v8h;
typedef __attribute__((ext_vector_type(8)))  float    v8f;

// ---------------------------------------------------------------------------
// Kernel 0: f32 -> f16 conversion (x and the three weight matrices)
// ---------------------------------------------------------------------------
__global__ void cvt_f32_f16(const float* __restrict__ src,
                            _Float16* __restrict__ dst, int n) {
  int i = blockIdx.x * blockDim.x + threadIdx.x;
  if (i < n) dst[i] = (_Float16)src[i];
}

// ---------------------------------------------------------------------------
// Kernel 1: Q/K/V = x @ W.T + b via v_wmma_f32_16x16x32_f16
// grid.x = 64 blocks (8 waves each -> 512 16x16 tiles), grid.y = 3 (Q,K,V)
// A layout (16-bit A 16x32, ISA 7.12.2): lane<16 row M=lane holds K0-7,K16-23;
//   lane>=16 row M=lane-16 holds K8-15,K24-31.
// B layout (32x16): lane<16 col N=lane holds K0-7,K16-23; lane>=16 col N=lane-16
//   holds K8-15,K24-31.
// ---------------------------------------------------------------------------
__global__ __launch_bounds__(256) void qkv_gemm(
    const _Float16* __restrict__ xh, const _Float16* __restrict__ Wh,
    const float* __restrict__ bqp, const float* __restrict__ bkp,
    const float* __restrict__ bvp,
    float* __restrict__ Qp, float* __restrict__ Kp, float* __restrict__ Vp) {
  const int which = blockIdx.y;
  const _Float16* W = Wh + (size_t)which * kD * kD;
  const float* bias = (which == 0) ? bqp : (which == 1) ? bkp : bvp;
  float* outp       = (which == 0) ? Qp  : (which == 1) ? Kp  : Vp;

  const int wave = threadIdx.x >> 5;
  const int lane = threadIdx.x & 31;
  const int tile = blockIdx.x * 8 + wave;   // 0..511
  const int m0 = (tile & 7) << 4;           // 8 M-tiles (B=128)
  const int n0 = (tile >> 3) << 4;          // 64 N-tiles (D=1024)
  const int lr = lane & 15;
  const int hi = lane >> 4;

  const _Float16* arow = xh + (size_t)(m0 + lr) * kD;
  const _Float16* brow = W  + (size_t)(n0 + lr) * kD;

  v8f c = {};
  for (int k0 = 0; k0 < kD; k0 += 32) {
    v8h a0 = *(const v8h*)(arow + k0 + hi * 8);
    v8h a1 = *(const v8h*)(arow + k0 + 16 + hi * 8);
    v8h b0 = *(const v8h*)(brow + k0 + hi * 8);
    v8h b1 = *(const v8h*)(brow + k0 + 16 + hi * 8);
    v16h a  = __builtin_shufflevector(a0, a1, 0,1,2,3,4,5,6,7,8,9,10,11,12,13,14,15);
    v16h bm = __builtin_shufflevector(b0, b1, 0,1,2,3,4,5,6,7,8,9,10,11,12,13,14,15);
    // D = A(16x32,f16) x B(32x16,f16) + C(f32)
    c = __builtin_amdgcn_wmma_f32_16x16x32_f16(
        /*neg_a=*/false, a, /*neg_b=*/false, bm,
        /*c_mod=*/(short)0, c, /*reuse_a=*/false, /*reuse_b=*/false);
  }

  // C/D layout: lane<16 -> rows m0+0..7, col n0+lane; lane>=16 -> rows m0+8..15
  const float bb = bias[n0 + lr];
  float* orow = outp + (size_t)(m0 + hi * 8) * kD + (n0 + lr);
#pragma unroll
  for (int r = 0; r < 8; ++r) orow[(size_t)r * kD] = c[r] + bb;
}

// ---------------------------------------------------------------------------
// Kernel 2: fused rank-1 softmax + attn materialization + out = attn @ V
// grid = (16, 128): block handles 64 rows of one batch; 8 waves x 8 rows.
// Each wave owns one row: e-row lives in 32 registers (wave32), reductions
// via __shfl_xor, attn streamed with nontemporal stores (512 MB, L2-bypass).
// ---------------------------------------------------------------------------
__global__ __launch_bounds__(256) void feat_attn(
    const float* __restrict__ Qp, const float* __restrict__ Kp,
    const float* __restrict__ Vp, float* __restrict__ outp,
    float* __restrict__ attn) {
  __shared__ float Ks[kD];
  __shared__ float Vs[kD];
  __shared__ float redmax[256];
  __shared__ float redmin[256];

  const int b = blockIdx.y;
  const int rowBase = blockIdx.x * 64;
  const int tid  = threadIdx.x;
  const int wave = tid >> 5;
  const int lane = tid & 31;

  for (int t = tid; t < kD; t += 256) {
    Ks[t] = Kp[(size_t)b * kD + t];
    Vs[t] = Vp[(size_t)b * kD + t];
  }
  __syncthreads();

  // Per-batch max/min of K: row max of q*K is q*maxK (q>=0) or q*minK (q<0)
  float mx = -3.402823466e38f, mn = 3.402823466e38f;
  for (int t = tid; t < kD; t += 256) {
    float k = Ks[t];
    mx = fmaxf(mx, k);
    mn = fminf(mn, k);
  }
  redmax[tid] = mx; redmin[tid] = mn;
  __syncthreads();
  for (int s = 128; s > 0; s >>= 1) {
    if (tid < s) {
      redmax[tid] = fmaxf(redmax[tid], redmax[tid + s]);
      redmin[tid] = fminf(redmin[tid], redmin[tid + s]);
    }
    __syncthreads();
  }
  const float maxK = redmax[0], minK = redmin[0];
  const float scale = 0.03125f;  // 1/sqrt(1024)

  for (int r = 0; r < 8; ++r) {
    const int i = rowBase + r * 8 + wave;
    const float qs = Qp[(size_t)b * kD + i] * scale;
    const float m  = (qs >= 0.f) ? qs * maxK : qs * minK;  // exact row max

    float e[32];
    float sum = 0.f, sev = 0.f;
#pragma unroll
    for (int jj = 0; jj < 32; ++jj) {
      const int j = lane + (jj << 5);
      float ee = __expf(qs * Ks[j] - m);
      e[jj] = ee;
      sum += ee;
      sev += ee * Vs[j];
    }
#pragma unroll
    for (int off = 16; off > 0; off >>= 1) {
      sum += __shfl_xor(sum, off, 32);
      sev += __shfl_xor(sev, off, 32);
    }
    const float inv = 1.0f / sum;
    float* arow = attn + ((size_t)b * kD + (size_t)i) * kD;
#pragma unroll
    for (int jj = 0; jj < 32; ++jj)
      __builtin_nontemporal_store(e[jj] * inv, arow + lane + (jj << 5));
    if (lane == 0) outp[(size_t)b * kD + i] = sev * inv;
  }
}

// ---------------------------------------------------------------------------
extern "C" void kernel_launch(void* const* d_in, const int* in_sizes, int n_in,
                              void* d_out, int out_size, void* d_ws,
                              size_t ws_size, hipStream_t stream) {
  (void)in_sizes; (void)n_in; (void)out_size; (void)ws_size;

  const float* x  = (const float*)d_in[0];
  const float* Wq = (const float*)d_in[1];
  const float* bq = (const float*)d_in[2];
  const float* Wk = (const float*)d_in[3];
  const float* bk = (const float*)d_in[4];
  const float* Wv = (const float*)d_in[5];
  const float* bv = (const float*)d_in[6];

  // Workspace layout (all offsets 256B-aligned):
  //   [0)        Q f32  (512 KB)
  //   [512K)     K f32  (512 KB)
  //   [1M)       V f32  (512 KB)
  //   [1.5M)     x  f16 (256 KB)
  //   [1.75M)    Wq|Wk|Wv f16 (3 x 2 MB)
  char* ws = (char*)d_ws;
  const size_t QKV_BYTES = (size_t)kB * kD * sizeof(float);
  float*    Qp = (float*)(ws);
  float*    Kp = (float*)(ws + QKV_BYTES);
  float*    Vp = (float*)(ws + 2 * QKV_BYTES);
  _Float16* xh = (_Float16*)(ws + 3 * QKV_BYTES);
  _Float16* Wh = (_Float16*)(ws + 3 * QKV_BYTES + (size_t)kB * kD * 2);

  float* outp = (float*)d_out;                 // [B, D]
  float* attn = outp + (size_t)kB * kD;        // [B, D, D]

  const int nX = kB * kD, nW = kD * kD;
  cvt_f32_f16<<<(nX + 255) / 256, 256, 0, stream>>>(x, xh, nX);
  cvt_f32_f16<<<(nW + 255) / 256, 256, 0, stream>>>(Wq, Wh, nW);
  cvt_f32_f16<<<(nW + 255) / 256, 256, 0, stream>>>(Wk, Wh + (size_t)nW, nW);
  cvt_f32_f16<<<(nW + 255) / 256, 256, 0, stream>>>(Wv, Wh + 2 * (size_t)nW, nW);

  qkv_gemm<<<dim3(64, 3), 256, 0, stream>>>(xh, Wh, bq, bk, bv, Qp, Kp, Vp);

  feat_attn<<<dim3(kD / 64, kB), 256, 0, stream>>>(Qp, Kp, Vp, outp, attn);
}
